// NeighborEmbedding_52561809768953
// MI455X (gfx1250) — matrix operand order repaired and checked by reading, not compiled, not run on hardware
//
#include <hip/hip_runtime.h>
#include <hip/hip_bf16.h>

// NeighborEmbedding for MI455X (gfx1250, wave32):
//  - edge GEMM (26 GFLOP) + node GEMM via v_wmma_f32_16x16x32_bf16, f32 accum
//  - fused cutoff-scale * emb gather * global_atomic_add_f32 scatter epilogue
//  - roofline: ~0.5 GB traffic @ 23.3 TB/s -> bf16 WMMA keeps it memory-bound

#define HIDDEN   128
#define NUM_RBF  64
#define CUTOFF_F 5.0f
#define PI_OVER_CUT 0.62831853071795864769f  // pi / 5

typedef __attribute__((ext_vector_type(16))) __bf16 v16bf;
typedef __attribute__((ext_vector_type(8)))  __bf16 v8bf;
typedef __attribute__((ext_vector_type(8)))  float  v8f;

// Build a bf16 16-element A operand fragment for this lane:
// elements 0..7 from 8 contiguous f32 at p0, elements 8..15 from p1.
__device__ __forceinline__ v16bf cvt16(const float* __restrict__ p0,
                                       const float* __restrict__ p1) {
  float4 a0 = *(const float4*)(p0);
  float4 a1 = *(const float4*)(p0 + 4);
  float4 b0 = *(const float4*)(p1);
  float4 b1 = *(const float4*)(p1 + 4);
  v16bf r;
  r[0]  = (__bf16)a0.x;  r[1]  = (__bf16)a0.y;
  r[2]  = (__bf16)a0.z;  r[3]  = (__bf16)a0.w;
  r[4]  = (__bf16)a1.x;  r[5]  = (__bf16)a1.y;
  r[6]  = (__bf16)a1.z;  r[7]  = (__bf16)a1.w;
  r[8]  = (__bf16)b0.x;  r[9]  = (__bf16)b0.y;
  r[10] = (__bf16)b0.z;  r[11] = (__bf16)b0.w;
  r[12] = (__bf16)b1.x;  r[13] = (__bf16)b1.y;
  r[14] = (__bf16)b1.z;  r[15] = (__bf16)b1.w;
  return r;
}

// Load 16 contiguous bf16 (B operand fragment: lane holds K=half*16+0..15 of its column)
__device__ __forceinline__ v16bf loadb16(const __bf16* __restrict__ p) {
  v8bf lo = *(const v8bf*)(p);
  v8bf hi = *(const v8bf*)(p + 8);
  return __builtin_shufflevector(lo, hi, 0,1,2,3,4,5,6,7,8,9,10,11,12,13,14,15);
}

// ---------------- prep kernels ----------------

__global__ void zero_f4_kernel(float4* __restrict__ p, int n4) {
  int i = blockIdx.x * blockDim.x + threadIdx.x;
  float4 z = make_float4(0.f, 0.f, 0.f, 0.f);
  for (; i < n4; i += gridDim.x * blockDim.x) p[i] = z;
}

__global__ void edge_scale_kernel(const int* __restrict__ src,
                                  const int* __restrict__ dst,
                                  const float* __restrict__ w,
                                  float* __restrict__ sE, int E) {
  int e = blockIdx.x * blockDim.x + threadIdx.x;
  if (e >= E) return;
  float wgt = w[e];
  float C = 0.5f * (__cosf(wgt * PI_OVER_CUT) + 1.0f);
  C = (wgt < CUTOFF_F) ? C : 0.f;
  sE[e] = (src[e] != dst[e]) ? C : 0.f;
}

__global__ void cvt_bf16_kernel(const float* __restrict__ a,
                                __bf16* __restrict__ b, int n) {
  int i = blockIdx.x * blockDim.x + threadIdx.x;
  for (; i < n; i += gridDim.x * blockDim.x) b[i] = (__bf16)a[i];
}

// ---------------- edge message + scatter kernel ----------------
// one wave32 per 16-edge tile: W[16x128] = edge_attr[16x64] @ projW^T via
// 8 h-tiles x 2 bf16 WMMAs, fused epilogue scatters with f32 atomics.

__global__ void __launch_bounds__(128)
edge_msg_kernel(const float* __restrict__ edge_attr,
                const int*   __restrict__ src,
                const int*   __restrict__ dst,
                const float* __restrict__ sE,
                const __bf16* __restrict__ projb16,
                const float* __restrict__ proj_b,
                const float* __restrict__ emb_W,
                const int*   __restrict__ z,
                float* __restrict__ agg,
                int E) {
  const int lane  = threadIdx.x & 31;
  const int wave  = threadIdx.x >> 5;
  const int etile = blockIdx.x * 4 + wave;
  const int e0    = etile * 16;
  if (e0 + 16 > E) return;                 // wave-uniform: EXEC stays all-1 for WMMA

  const int n    = lane & 15;
  const int half = lane >> 4;

  // A operands: lane holds row e0+n; K = half*16+0..15 (contiguous, matches B)
  const float* arow = edge_attr + (size_t)(e0 + n) * NUM_RBF;
  v16bf a0 = cvt16(arow + half * 16,      arow + half * 16 + 8);   // K = 0..31
  v16bf a1 = cvt16(arow + 32 + half * 16, arow + 40 + half * 16);   // K = 32..63

  // prefetch a future tile's A data into cache
  __builtin_prefetch(edge_attr + (size_t)(e0 + 256 + n) * NUM_RBF, 0, 1);

  // per-row (epilogue) metadata: C/D row M = r + 8*half
  float sv[8]; int sb[8]; int zr[8];
#pragma unroll
  for (int r = 0; r < 8; ++r) {
    const int e = e0 + r + 8 * half;
    sv[r] = sE[e];
    sb[r] = src[e] * HIDDEN;
    zr[r] = z[dst[e]] * HIDDEN;
  }

#pragma unroll
  for (int h = 0; h < 8; ++h) {
    const int col = h * 16 + n;
    const __bf16* brow = projb16 + (size_t)col * NUM_RBF;
    v16bf b0 = loadb16(brow + half * 16);        // K = 0..31 of column `col`
    v16bf b1 = loadb16(brow + 32 + half * 16);   // K = 32..63
    v8f c = {0.f, 0.f, 0.f, 0.f, 0.f, 0.f, 0.f, 0.f};
    c = __builtin_amdgcn_wmma_f32_16x16x32_bf16(false, a0, false, b0,
                                                (short)0, c, false, false);
    c = __builtin_amdgcn_wmma_f32_16x16x32_bf16(false, a1, false, b1,
                                                (short)0, c, false, false);
    const float pb = proj_b[col];
#pragma unroll
    for (int r = 0; r < 8; ++r) {
      const float wv = (c[r] + pb) * sv[r];
      const float em = emb_W[zr[r] + col];
      __hip_atomic_fetch_add(&agg[sb[r] + col], wv * em,
                             __ATOMIC_RELAXED, __HIP_MEMORY_SCOPE_AGENT);
    }
  }
}

// ---------------- final combine kernel ----------------
// out[16x16 tile] = concat(x,agg)[16x256] @ combW^T tile, K=256 as 8 bf16 WMMAs

__global__ void __launch_bounds__(128)
combine_kernel(const float* __restrict__ x,
               const float* __restrict__ agg,
               const __bf16* __restrict__ combb16,
               const float* __restrict__ comb_b,
               float* __restrict__ out,
               int Nnodes) {
  const int lane  = threadIdx.x & 31;
  const int wave  = threadIdx.x >> 5;
  const int wid   = blockIdx.x * 4 + wave;
  const int ntile = wid >> 3;
  const int htile = wid & 7;
  const int n0    = ntile * 16;
  if (n0 + 16 > Nnodes) return;            // wave-uniform

  const int n    = lane & 15;
  const int half = lane >> 4;

  const float*  xrow = x   + (size_t)(n0 + n) * HIDDEN;
  const float*  grow = agg + (size_t)(n0 + n) * HIDDEN;
  const __bf16* brow = combb16 + (size_t)(htile * 16 + n) * (2 * HIDDEN);

  v8f c = {0.f, 0.f, 0.f, 0.f, 0.f, 0.f, 0.f, 0.f};
#pragma unroll
  for (int kt = 0; kt < 8; ++kt) {
    const float* ap = (kt < 4) ? (xrow + kt * 32) : (grow + (kt - 4) * 32);
    v16bf a = cvt16(ap + half * 16, ap + half * 16 + 8);
    v16bf b = loadb16(brow + kt * 32 + half * 16);
    c = __builtin_amdgcn_wmma_f32_16x16x32_bf16(false, a, false, b,
                                                (short)0, c, false, false);
  }
  const float bias = comb_b[htile * 16 + n];
#pragma unroll
  for (int r = 0; r < 8; ++r) {
    out[(size_t)(n0 + r + 8 * half) * HIDDEN + htile * 16 + n] = c[r] + bias;
  }
}

// ---------------- launcher ----------------

extern "C" void kernel_launch(void* const* d_in, const int* in_sizes, int n_in,
                              void* d_out, int out_size, void* d_ws, size_t ws_size,
                              hipStream_t stream) {
  const int*   z     = (const int*)  d_in[0];
  const float* x     = (const float*)d_in[1];
  const int*   eidx  = (const int*)  d_in[2];   // [2, E]: src row then dst row
  const float* ew    = (const float*)d_in[3];
  const float* eattr = (const float*)d_in[4];
  const float* embW  = (const float*)d_in[5];
  const float* projW = (const float*)d_in[6];   // [128, 64]
  const float* projb = (const float*)d_in[7];
  const float* combW = (const float*)d_in[8];   // [128, 256]
  const float* combb = (const float*)d_in[9];
  float* out = (float*)d_out;

  const int N = in_sizes[0];
  const int E = in_sizes[3];
  const int* srcI = eidx;
  const int* dstI = eidx + E;

  // workspace layout (all offsets 16B aligned for the given N, E)
  char* ws = (char*)d_ws;
  float*  agg     = (float*)ws;                                   // N*128 f32 (zeroed)
  float*  sEdge   = (float*)(ws + (size_t)N * HIDDEN * 4);        // E f32
  __bf16* projb16 = (__bf16*)(ws + (size_t)N * HIDDEN * 4 + (size_t)E * 4);
  __bf16* combb16 = (__bf16*)((char*)projb16 + (size_t)HIDDEN * NUM_RBF * 2);

  // 1) zero the accumulator (fresh every call -> deterministic replay)
  {
    int n4 = (N * HIDDEN) / 4;
    zero_f4_kernel<<<(n4 + 255) / 256, 256, 0, stream>>>((float4*)agg, n4);
  }
  // 2) per-edge cutoff * self-loop-mask scale
  edge_scale_kernel<<<(E + 255) / 256, 256, 0, stream>>>(srcI, dstI, ew, sEdge, E);
  // 3) bf16 copies of the two weight matrices (tiny, L2-resident)
  cvt_bf16_kernel<<<32, 256, 0, stream>>>(projW, projb16, HIDDEN * NUM_RBF);
  cvt_bf16_kernel<<<128, 256, 0, stream>>>(combW, combb16, HIDDEN * 2 * HIDDEN);

  // 4) edge GEMM + fused scale/gather/atomic-scatter
  {
    int etiles = (E + 15) / 16;
    int blocks = (etiles + 3) / 4;            // 4 waves / block
    edge_msg_kernel<<<blocks, 128, 0, stream>>>(eattr, srcI, dstI, sEdge,
                                                projb16, projb, embW, z, agg, E);
  }
  // 5) final combine GEMM -> out
  {
    int ntiles = (N + 15) / 16;
    int waves  = ntiles * 8;                  // 8 h-tiles per node tile
    int blocks = (waves + 3) / 4;
    combine_kernel<<<blocks, 128, 0, stream>>>(x, agg, combb16, combb, out, N);
  }
}
